// Graph_Editer_30623116821143
// MI455X (gfx1250) — compile-verified, matrix operand order checked by reading, not run on hardware
//
#include <hip/hip_runtime.h>
#include <cmath>
#include <cstdint>

typedef __attribute__((ext_vector_type(16))) _Float16 v16h;
typedef __attribute__((ext_vector_type(8)))  float    v8f;

#define MAXN 5120          // capacity for one row of keys in LDS (n = 5003)
#define TPB  256
#define MAXIT ((MAXN + TPB - 1) / TPB)   // 20 strided iterations per thread

// ---------------------------------------------------------------- utilities
__device__ __forceinline__ uint32_t rotl32(uint32_t x, uint32_t r) {
    return (x << r) | (x >> (32u - r));
}

// Threefry-2x32 (20 rounds), key = (0, 42), counter = flat element index.
// Returns a Gumbel(0,1) sample.
__device__ __forceinline__ float gumbel_rng(uint32_t c_hi, uint32_t c_lo) {
    const uint32_t k0 = 0u, k1 = 42u;
    const uint32_t k2 = 0x1BD11BDAu ^ k0 ^ k1;
    const uint32_t ks[3] = {k0, k1, k2};
    const uint32_t R[8] = {13u, 15u, 26u, 6u, 17u, 29u, 16u, 24u};
    uint32_t x0 = c_hi + k0, x1 = c_lo + k1;
#pragma unroll
    for (int blk = 0; blk < 5; ++blk) {
#pragma unroll
        for (int rr = 0; rr < 4; ++rr) {
            uint32_t rot = R[(blk & 1) * 4 + rr];
            x0 += x1; x1 = rotl32(x1, rot); x1 ^= x0;
        }
        x0 += ks[(blk + 1) % 3];
        x1 += ks[(blk + 2) % 3] + (uint32_t)(blk + 1);
    }
    uint32_t b = x0 ^ x1;
    float u = (float)(b >> 8) * 0x1.0p-24f + 0x1.0p-25f;  // (0,1)
    return -__logf(-__logf(u));
}

// ---------------------------------------------------------------- kernels
__global__ void init_kernel(float* __restrict__ out, size_t total,
                            float* __restrict__ wsf, int nws) {
    size_t t = (size_t)blockIdx.x * TPB + threadIdx.x;
    if (t < total) out[t] = 0.0f;
    if (t < (size_t)nws) wsf[t] = 0.0f;
}

// Column max of Bk via integer atomicMax on float bits (valid: B >= 0).
__global__ void colmax_kernel(const float* __restrict__ B, const int* __restrict__ kptr,
                              float* __restrict__ colmax, int n) {
    int c = blockIdx.x * TPB + threadIdx.x;
    if (c >= n) return;
    const float* Bk = B + (size_t)(*kptr) * (size_t)n * (size_t)n;
    int nchunks = gridDim.y, chunk = blockIdx.y;
    int r0 = (int)(((long long)chunk * n) / nchunks);
    int r1 = (int)(((long long)(chunk + 1) * n) / nchunks);
    float m = 0.0f;  // B ~ U(0,1) >= 0
    const float* p = Bk + (size_t)r0 * (size_t)n + c;
    for (int r = r0; r < r1; ++r) { m = fmaxf(m, *p); p += n; }
    atomicMax((int*)&colmax[c], __float_as_int(m));
}

// logZ[c] = colmax[c] + log(sum_r exp(Bk[r,c]-colmax[c])).
// Column sums via D += ones(16x32,f16) x E(32x16,f16) with
// v_wmma_f32_16x16x32_f16 (f32 accumulate). One 256-thread block per
// 16-column strip; 8 waves split the rows, combine via LDS.
// Branchless: column index clamped (garbage cols dropped at the guarded
// store); only the single wave-uniform tail chunk masks *values*.
__global__ void logz_kernel(const float* __restrict__ B, const int* __restrict__ kptr,
                            const float* __restrict__ colmax,
                            float* __restrict__ logZ, float* __restrict__ logp, int n) {
    __shared__ float part[16];
    int tid = threadIdx.x;
    int wave = tid >> 5, lane = tid & 31;
    int cbase = blockIdx.x * 16;
    int col = cbase + (lane & 15);
    int colc = col < n ? col : n - 1;            // clamp: loads always legal
    const size_t ns = (size_t)n;
    const float* Bk = B + (size_t)(*kptr) * ns * ns;
    float m = colmax[colc];

    v16h a;
#pragma unroll
    for (int i = 0; i < 16; ++i) a[i] = (_Float16)1.0f;
    v8f acc = {};

    int rowhalf = (lane >> 4) << 4;              // lanes 16-31 take rows +16
    int nfull = n >> 5;                          // chunks with all 32 rows valid

    for (int chunk = wave; chunk < nfull; chunk += 8) {
        const float* p = Bk + ((size_t)(chunk << 5) + (size_t)rowhalf) * ns + colc;
        v16h bm;
#pragma unroll
        for (int i = 0; i < 16; ++i)
            bm[i] = (_Float16)__expf(p[(size_t)i * ns] - m);   // in (0,1]
        acc = __builtin_amdgcn_wmma_f32_16x16x32_f16(
            false, a, false, bm, (short)0, acc, false, false);
    }

    // Wave-uniform tail chunk: clamp row address, mask the value only.
    if ((n & 31) && wave == (nfull & 7)) {
        int rbase = (nfull << 5) + rowhalf;
        v16h bm;
#pragma unroll
        for (int i = 0; i < 16; ++i) {
            int rr = rbase + i;
            int rc = rr < n ? rr : n - 1;
            float e = __expf(Bk[(size_t)rc * ns + colc] - m);
            bm[i] = (_Float16)(rr < n ? e : 0.0f);
        }
        acc = __builtin_amdgcn_wmma_f32_16x16x32_f16(
            false, a, false, bm, (short)0, acc, false, false);
    }

    if (tid < 16) part[tid] = 0.0f;
    __syncthreads();
    if (lane < 16) atomicAdd(&part[lane], acc[0]);   // D[0][lane]
    __syncthreads();
    if (tid < 16) {
        int c = cbase + tid;
        if (c < n) {
            float lz = colmax[c] + __logf(part[tid]);
            logZ[c] = lz;
            atomicAdd(logp, -lz);   // log_p -= sum(logZ)
        }
    }
}

// A[src,dst] = 1 written directly into C (duplicates benign).
__global__ void scatterA_kernel(const int* __restrict__ edge, int n_edges, int n,
                                float* __restrict__ C) {
    int e = blockIdx.x * TPB + threadIdx.x;
    if (e >= n_edges) return;
    int src = edge[e], dst = edge[n_edges + e];
    C[(size_t)src * (size_t)n + (size_t)dst] = 1.0f;
}

// One block per row r. Async-DMA Bk[r,:] into LDS, overlap Threefry/gumbel
// with the DMA, then num_sample argmax-and-knockout rounds.
// Reduction: 5-step wave shuffle (no barriers) + 8 wave winners in LDS.
__global__ void topk_kernel(const float* __restrict__ B, const int* __restrict__ kptr,
                            const float* __restrict__ logZ, const int* __restrict__ nsptr,
                            int* __restrict__ S, int n) {
    __shared__ float keybuf[MAXN];
    __shared__ float wredv[8];
    __shared__ int   wredi[8];
    __shared__ int   winner;

    int r = blockIdx.x, tid = threadIdx.x;
    int wave = tid >> 5, lane = tid & 31;
    int ns = *nsptr;
    const float* Bk = B + (size_t)(*kptr) * (size_t)n * (size_t)n;
    const float* rowbase = Bk + (size_t)r * (size_t)n;  // wave-uniform SGPR base

    // Async global->LDS copies of this row (no VGPR data path, ASYNCcnt).
    for (int c = tid; c < n; c += TPB) {
        uint32_t lds_addr = (uint32_t)(uintptr_t)&keybuf[c];
        asm volatile("global_load_async_to_lds_b32 %0, %1, %2"
                     :: "v"(lds_addr), "v"(c * 4), "s"(rowbase)
                     : "memory");
    }

    // Overlap: gumbel - logZ while the DMA is in flight.
    float gum[MAXIT];
#pragma unroll
    for (int it = 0; it < MAXIT; ++it) {
        int c = tid + it * TPB;
        if (c < n) {
            uint32_t idx = (uint32_t)r * (uint32_t)n + (uint32_t)c;
            gum[it] = gumbel_rng(0u, idx) - logZ[c];
        }
    }

    asm volatile("s_wait_asynccnt 0" ::: "memory");
    __syncthreads();

#pragma unroll
    for (int it = 0; it < MAXIT; ++it) {
        int c = tid + it * TPB;
        if (c < n) keybuf[c] += gum[it];
    }
    __syncthreads();

    for (int j = 0; j < ns; ++j) {
        float bv = -INFINITY; int bi = 0;
        for (int c = tid; c < n; c += TPB) {
            float v = keybuf[c];
            if (v > bv) { bv = v; bi = c; }
        }
        // wave32 shuffle argmax: no barriers inside the wave
#pragma unroll
        for (int off = 16; off > 0; off >>= 1) {
            float ov = __shfl_down(bv, off);
            int   oi = __shfl_down(bi, off);
            if (ov > bv) { bv = ov; bi = oi; }
        }
        if (lane == 0) { wredv[wave] = bv; wredi[wave] = bi; }
        __syncthreads();
        if (tid == 0) {
            float fbv = wredv[0]; int fbi = wredi[0];
#pragma unroll
            for (int w = 1; w < 8; ++w)
                if (wredv[w] > fbv) { fbv = wredv[w]; fbi = wredi[w]; }
            S[(size_t)r * (size_t)ns + j] = fbi;
            keybuf[fbi] = -INFINITY;   // knock out
            winner = fbi;
        }
        __syncthreads();
        (void)winner;
    }
}

// Flip sampled positions (C = A xor M; positions unique -> race-free) and
// accumulate log_p += sum Bk[S[i,j], i].
__global__ void flip_logp_kernel(const float* __restrict__ B, const int* __restrict__ kptr,
                                 const int* __restrict__ S, const int* __restrict__ nsptr,
                                 float* __restrict__ C, float* __restrict__ logp, int n) {
    int i = blockIdx.x, tid = threadIdx.x;
    int lane = tid & 31, wave = tid >> 5;
    __shared__ float wsum[4];
    int ns = *nsptr;
    const float* Bk = B + (size_t)(*kptr) * (size_t)n * (size_t)n;
    float local = 0.0f;
    for (int j = tid; j < ns; j += 128) {
        int s = S[(size_t)i * (size_t)ns + j];
        size_t idx = (size_t)s * (size_t)n + (size_t)i;
        C[idx] = 1.0f - C[idx];
        local += Bk[idx];
    }
#pragma unroll
    for (int off = 16; off > 0; off >>= 1) local += __shfl_down(local, off);
    if (lane == 0) wsum[wave] = local;
    __syncthreads();
    if (tid == 0)
        atomicAdd(logp, wsum[0] + wsum[1] + wsum[2] + wsum[3]);
}

// ---------------------------------------------------------------- launch
extern "C" void kernel_launch(void* const* d_in, const int* in_sizes, int n_in,
                              void* d_out, int out_size, void* d_ws, size_t ws_size,
                              hipStream_t stream) {
    const int*   edge = (const int*)d_in[0];
    const float* B    = (const float*)d_in[1];
    // d_in[2] = n (device scalar, unused: derived from out_size)
    const int*   d_ns = (const int*)d_in[3];   // num_sample (device scalar)
    const int*   d_k  = (const int*)d_in[4];   // k          (device scalar)

    int n       = (int)(sqrt((double)(out_size - 1)) + 0.5);  // out = n*n + 1
    int n_edges = in_sizes[0] / 2;

    float* C    = (float*)d_out;
    float* logp = C + (size_t)n * (size_t)n;

    float* wsf    = (float*)d_ws;
    float* logZ   = wsf;                          // n floats
    float* colmax = wsf + n;                      // n floats
    int*   S      = (int*)(wsf + 2 * (size_t)n);  // n * num_sample ints

    size_t total = (size_t)n * (size_t)n + 1;

    init_kernel<<<(unsigned)((total + TPB - 1) / TPB), TPB, 0, stream>>>(
        C, total, wsf, 2 * n);

    dim3 gcm((n + TPB - 1) / TPB, 40);
    colmax_kernel<<<gcm, TPB, 0, stream>>>(B, d_k, colmax, n);

    logz_kernel<<<(n + 15) / 16, TPB, 0, stream>>>(B, d_k, colmax, logZ, logp, n);

    scatterA_kernel<<<(n_edges + TPB - 1) / TPB, TPB, 0, stream>>>(edge, n_edges, n, C);

    topk_kernel<<<n, TPB, 0, stream>>>(B, d_k, logZ, d_ns, S, n);

    flip_logp_kernel<<<n, 128, 0, stream>>>(B, d_k, S, d_ns, C, logp, n);
}
